// RGCNBlock_50242527428805
// MI455X (gfx1250) — matrix operand order repaired and checked by reading, not compiled, not run on hardware
//
#include <hip/hip_runtime.h>

typedef __bf16 bf16;
typedef __attribute__((ext_vector_type(16))) __bf16 v16bf;
typedef __attribute__((ext_vector_type(8)))  float  v8f;
typedef __attribute__((ext_vector_type(4)))  unsigned int v4u;
typedef __attribute__((ext_vector_type(8)))  int v8i;
typedef __attribute__((ext_vector_type(4)))  int v4i;

#define Bsz 512
#define Csz 256
#define Rsz 8
#define Nsz 81
#define NP  96          // padded node count (multiple of 32)
#define SH  264         // LDS row stride for h (256 + 8)
#define SA  104         // LDS row stride for adj (96 + 8)
#define PW  520         // packed B-operand row stride (512 + 8) -> 1024B + 16B pad
#define TPB 512         // 16 wave32 per block

#if __has_builtin(__builtin_amdgcn_tensor_load_to_lds) && __has_builtin(__builtin_amdgcn_s_wait_tensorcnt)
#define HAVE_TDM 1
#else
#define HAVE_TDM 0
#endif

__device__ inline v8f v8f_zero() {
  v8f z;
#pragma unroll
  for (int i = 0; i < 8; ++i) z[i] = 0.0f;
  return z;
}

// A-fragment: 16x32 bf16 tile, row-major LDS (stride in elements).
// Lanes 0-15: M=lane, K 0..15; lanes 16-31: same M, K 16..31.
__device__ inline v16bf load_a_frag(const bf16* sh, int m0, int k0, int stride, int lane) {
  v16bf a;
  const int m  = m0 + (lane & 15);
  const int kh = (lane >> 4) << 4;
  const bf16* row = sh + m * stride + k0 + kh;
#pragma unroll
  for (int v = 0; v < 8; ++v) {
    const int k = 2 * v;
    a[2 * v]     = row[k];
    a[2 * v + 1] = row[k + 1];
  }
  return a;
}

// B-fragment from K-pair-packed LDS: element (k,n) lives at
// (k>>1)*pstride + 2n + (k&1).  Each lane reads 8 aligned b32 words.
__device__ inline v16bf load_b_frag_packed(const bf16* sh, int k0, int n0, int pstride, int lane) {
  union { v16bf v; unsigned int u[8]; } f;
  const int n  = n0 + (lane & 15);
  const int kb = (lane >> 4) << 4;
  const bf16* base = sh + ((k0 + kb) >> 1) * pstride + n * 2;
#pragma unroll
  for (int v = 0; v < 8; ++v)
    f.u[v] = *(const unsigned int*)(base + v * pstride);
  return f.v;
}

#if HAVE_TDM
// TDM: DMA one W chunk (16 packed rows x 512 bf16, contiguous in global) into
// LDS with row padding 1024B -> +16B (pad_interval=7: 256 DW, pad_amount=3: 4 DW).
__device__ inline void tdm_load_w_chunk(const bf16* gsrc, bf16* lds_dst) {
  unsigned long long ga = (unsigned long long)(uintptr_t)gsrc;
  unsigned int la = (unsigned int)(uintptr_t)lds_dst;  // flat LDS addr[31:0] == LDS offset
  v4u g0;
  g0.x = 1u;                                            // count=1 (valid user D#)
  g0.y = la;                                            // lds_addr
  g0.z = (unsigned int)ga;                              // global_addr low 32
  g0.w = (unsigned int)((ga >> 32) & 0x01FFFFFFull) | 0x80000000u;  // addr hi + type=2
  v8i g1;
  g1[0] = (int)((1u << 16) | (1u << 20) | (7u << 22) | (3u << 25));
  //        data_size=2B     pad_enable    pad_interval=7  pad_amount=3
  g1[1] = (int)(512u << 16);   // tensor_dim0 = 512 (bits 63:48)
  g1[2] = (int)(16u << 16);    // tensor_dim0 hi=0 | tensor_dim1 = 16 (bits 95:80)
  g1[3] = (int)(512u << 16);   // tensor_dim1 hi=0 | tile_dim0 = 512 (bits 127:112)
  g1[4] = 16;                  // tile_dim1 = 16, tile_dim2 = 0
  g1[5] = 512;                 // tensor_dim0_stride low 32
  g1[6] = 0;
  g1[7] = 0;
  v4i z4 = {0, 0, 0, 0};
  v8i z8;
#pragma unroll
  for (int i = 0; i < 8; ++i) z8[i] = 0;
  // 6-arg form (clang-23 / therock-10.0 headers): groups 2/3 unused (2D tile).
  __builtin_amdgcn_tensor_load_to_lds(g0, g1, z4, z4, z8, 0);
}
#endif

// ---------------- Kernel 0: weight fp32 -> bf16, K-pair-packed --------------
// wbf[r][c>>1][d][c&1]  (packed row = 512 elems = two K rows interleaved)
__global__ __launch_bounds__(256) void wconv_kernel(const float* __restrict__ w,
                                                    bf16* __restrict__ wb, int n) {
  int i = blockIdx.x * 256 + threadIdx.x;
  if (i < n) {
    int r = i >> 16, c = (i >> 8) & 255, d = i & 255;
    wb[(r << 16) + ((c >> 1) << 9) + (d << 1) + (c & 1)] = (bf16)w[i];
  }
}

// ---------------- Kernel 1: per-batch fused RGCN GEMMs ----------------------
__global__ __launch_bounds__(TPB, 1) void rgcn_gemm_kernel(
    const float* __restrict__ x, const float* __restrict__ adj,
    const bf16* __restrict__ wbf, float* __restrict__ pre,
    float* __restrict__ persum, float* __restrict__ persumsq) {
  __shared__ __align__(16) bf16 sh_h  [NP * SH];      // h[b], row-major M x K
  __shared__ __align__(16) bf16 sh_hw [48 * PW];      // hw_r, K-pair-packed (K=96)
  __shared__ __align__(16) bf16 sh_w  [2][16 * PW];   // W_r chunk, packed, dbl-buffered
  __shared__ __align__(16) bf16 sh_adj[NP * SA];      // adj_r padded 96x96

  const int b    = blockIdx.x;
  const int tid  = threadIdx.x;
  const int lane = tid & 31;
  const int wave = tid >> 5;      // 16 waves; wave == output col-tile
  const int half = lane >> 4;
  const int lcol = lane & 15;

  const float* xb = x + (size_t)b * (Csz * Nsz);

  // Stage h[b]: sh_h[m][c] = x[b][c][m], rows 81..95 zero.
  for (int i = tid; i < NP * Csz; i += TPB) {
    int m = i >> 8, c = i & 255;
    float v = (m < Nsz) ? xb[c * Nsz + m] : 0.0f;
    sh_h[m * SH + c] = (bf16)v;
  }
  // Zero adj buffer once; pad region stays zero for all relations.
  for (int i = tid; i < NP * SA; i += TPB) sh_adj[i] = (bf16)0.0f;

  v8f acc[6];
#pragma unroll
  for (int t = 0; t < 6; ++t) acc[t] = v8f_zero();

  __syncthreads();

  const int n0 = wave * 16;

  for (int r = 0; r < Rsz; ++r) {
    // Stage adj[b][r] (81x81) as bf16; prefetch next relation's slab.
    const float* ap = adj + (((size_t)b * Rsz + r) * Nsz) * Nsz;
    if (r + 1 < Rsz) {
      const char* pf = (const char*)(ap + Nsz * Nsz) + tid * 64;
      if (tid * 64 < Nsz * Nsz * 4) __builtin_prefetch(pf, 0, 1);
    }
    for (int i = tid; i < Nsz * Nsz; i += TPB) {
      int n = i / Nsz, m = i - n * Nsz;
      sh_adj[n * SA + m] = (bf16)ap[i];
    }

    v8f accA[6];
#pragma unroll
    for (int t = 0; t < 6; ++t) accA[t] = v8f_zero();

    const bf16* wr = wbf + ((size_t)r << 16);   // packed W_r (256x256 as 128x512)

#if HAVE_TDM
    if (wave == 0) tdm_load_w_chunk(wr, sh_w[0]);   // chunk 0, async
#endif
    // Stage A: hw_r = h @ W_r, K=256 streamed in 8 chunks of 32 (double-buffered).
    for (int kc = 0; kc < 8; ++kc) {
#if HAVE_TDM
      if (wave == 0) __builtin_amdgcn_s_wait_tensorcnt(0);
      __syncthreads();           // chunk kc visible; prev compute done
      if (wave == 0 && kc < 7)   // overlap DMA of chunk kc+1 with WMMAs of kc
        tdm_load_w_chunk(wr + (size_t)(kc + 1) * 8192, sh_w[(kc + 1) & 1]);
#else
      __syncthreads();
      {
        const bf16* src = wr + (size_t)kc * 8192;
        for (int i = tid; i < 1024; i += TPB) {   // 16 rows x 64 uint4
          int row = i >> 6, c8 = i & 63;
          *(uint4*)(&sh_w[kc & 1][row * PW + c8 * 8]) =
              *(const uint4*)(src + row * 512 + c8 * 8);
        }
      }
      __syncthreads();
#endif
      v16bf bfrag = load_b_frag_packed(sh_w[kc & 1], 0, n0, PW, lane);
#pragma unroll
      for (int mt = 0; mt < 6; ++mt) {
        v16bf afrag = load_a_frag(sh_h, mt * 16, kc * 32, SH, lane);
        accA[mt] = __builtin_amdgcn_wmma_f32_16x16x32_bf16(
            false, afrag, false, bfrag, (short)0, accA[mt], false, false);
      }
    }
    __syncthreads();

    // Scatter hw_r to K-pair-packed LDS. C/D layout: VGPR v -> M = v + 8*half.
#pragma unroll
    for (int mt = 0; mt < 6; ++mt) {
#pragma unroll
      for (int v = 0; v < 8; ++v) {
        int m = mt * 16 + v + half * 8;
        sh_hw[(m >> 1) * PW + (n0 + lcol) * 2 + (m & 1)] = (bf16)accA[mt][v];
      }
    }
    __syncthreads();

    // Stage B: acc += adj_r @ hw_r (K = 96 -> 3 WMMA steps).
#pragma unroll
    for (int ks = 0; ks < 3; ++ks) {
      v16bf bfrag = load_b_frag_packed(sh_hw, ks * 32, n0, PW, lane);
#pragma unroll
      for (int mt = 0; mt < 6; ++mt) {
        v16bf afrag = load_a_frag(sh_adj, mt * 16, ks * 32, SA, lane);
        acc[mt] = __builtin_amdgcn_wmma_f32_16x16x32_bf16(
            false, afrag, false, bfrag, (short)0, acc[mt], false, false);
      }
    }
    __syncthreads();  // protect sh_adj / sh_hw / sh_w before next relation
  }

  // Epilogue: residual add, emit pre (into d_out), per-(b,chan) stats.
  const int chan = n0 + lcol;
  float s1 = 0.0f, s2 = 0.0f;
  float* preb = pre + (size_t)b * (Csz * Nsz);
#pragma unroll
  for (int mt = 0; mt < 6; ++mt) {
#pragma unroll
    for (int v = 0; v < 8; ++v) {
      int m = mt * 16 + v + half * 8;
      if (m < Nsz) {
        float val = acc[mt][v] + xb[chan * Nsz + m];
        preb[chan * Nsz + m] = val;
        s1 += val;
        s2 += val * val;
      }
    }
  }
  s1 += __shfl_xor(s1, 16, 32);
  s2 += __shfl_xor(s2, 16, 32);
  if (half == 0) {
    persum  [b * Csz + chan] = s1;
    persumsq[b * Csz + chan] = s2;
  }
}

// ---------------- Kernel 2: deterministic BN stats reduction ----------------
__global__ __launch_bounds__(256) void bn_finalize_kernel(
    const float* __restrict__ persum, const float* __restrict__ persumsq,
    const float* __restrict__ gamma, const float* __restrict__ beta,
    float* __restrict__ ss) {
  int c = threadIdx.x;
  float s1 = 0.0f, s2 = 0.0f;
  for (int b = 0; b < Bsz; ++b) {
    s1 += persum[b * Csz + c];
    s2 += persumsq[b * Csz + c];
  }
  const float inv = 1.0f / ((float)Bsz * (float)Nsz);
  float mean = s1 * inv;
  float var  = s2 * inv - mean * mean;
  float sc   = gamma[c] * rsqrtf(var + 1e-5f);
  ss[c]       = sc;
  ss[Csz + c] = beta[c] - mean * sc;
}

// ---------------- Kernel 3: BN + ReLU in place on d_out ---------------------
__global__ __launch_bounds__(256) void bn_apply_kernel(float* __restrict__ out,
                                                       const float* __restrict__ ss,
                                                       int total) {
  int i = blockIdx.x * 256 + threadIdx.x;
  if (i < total) {
    int chan = (i / Nsz) & (Csz - 1);
    float v = out[i] * ss[chan] + ss[Csz + chan];
    out[i] = fmaxf(v, 0.0f);
  }
}

extern "C" void kernel_launch(void* const* d_in, const int* in_sizes, int n_in,
                              void* d_out, int out_size, void* d_ws, size_t ws_size,
                              hipStream_t stream) {
  const float* x     = (const float*)d_in[0];
  const float* adj   = (const float*)d_in[1];
  const float* wgt   = (const float*)d_in[2];
  const float* gamma = (const float*)d_in[3];
  const float* beta  = (const float*)d_in[4];
  float* out = (float*)d_out;

  char* ws = (char*)d_ws;
  bf16*  wbf      = (bf16*)ws;                                   // 1 MiB (packed)
  float* persum   = (float*)(ws + (1u << 20));                   // 512 KiB
  float* persumsq = (float*)(ws + (1u << 20) + (512u << 10));    // 512 KiB
  float* ss       = (float*)(ws + (2u << 20));                   // 2 KiB

  const int wn = Rsz * Csz * Csz;
  wconv_kernel<<<(wn + 255) / 256, 256, 0, stream>>>(wgt, wbf, wn);

  rgcn_gemm_kernel<<<Bsz, TPB, 0, stream>>>(x, adj, wbf, out, persum, persumsq);

  bn_finalize_kernel<<<1, Csz, 0, stream>>>(persum, persumsq, gamma, beta, ss);

  const int total = Bsz * Csz * Nsz;
  bn_apply_kernel<<<(total + 255) / 256, 256, 0, stream>>>(out, ss, total);
}
